// Attention_85392539779833
// MI455X (gfx1250) — matrix operand order repaired and checked by reading, not compiled, not run on hardware
//
#include <hip/hip_runtime.h>

// ---------------------------------------------------------------------------
// MI455X / gfx1250 fused attention:
//   LN -> bf16, QKV GEMM (pipelined WMMA bf16), flash attention (TDM-staged
//   K/V tiles in LDS, WMMA bf16, online softmax), out-proj GEMM + bias.
// ---------------------------------------------------------------------------

typedef __attribute__((ext_vector_type(16))) __bf16 bf16x16;
typedef __attribute__((ext_vector_type(8)))  __bf16 bf16x8;
typedef __attribute__((ext_vector_type(4)))  __bf16 bf16x4;
typedef __attribute__((ext_vector_type(8)))  float  f32x8;
typedef __attribute__((ext_vector_type(4)))  unsigned int u32x4;
typedef __attribute__((ext_vector_type(8)))  int i32x8;
typedef __attribute__((ext_vector_type(4)))  int i32x4;

#define N_TOK   2048
#define DIM     1024
#define HEADS   16
#define DHEAD   64

// ---- WMMA helpers ----------------------------------------------------------

__device__ __forceinline__ f32x8 wmma_bf16(bf16x16 a, bf16x16 b, f32x8 c) {
    return __builtin_amdgcn_wmma_f32_16x16x32_bf16(
        false, a, false, b, (short)0, c, false, false);
}

// A-matrix 16x32 bf16: lane m = lane%16, khalf = lane/16.
// VGPR0-3 hold K = khalf*8 + 0..7, VGPR4-7 hold K = 16 + khalf*8 + 0..7.
__device__ __forceinline__ bf16x16 load_a16x32(const __bf16* base, int ld) {
    const int l  = threadIdx.x & 31;
    const int m  = l & 15;
    const int kh = l >> 4;
    const __bf16* p = base + (size_t)m * ld + kh * 8;
    bf16x8 lo = *(const bf16x8*)(p);
    bf16x8 hi = *(const bf16x8*)(p + 16);
    bf16x16 r;
#pragma unroll
    for (int i = 0; i < 8; ++i) { r[i] = lo[i]; r[i + 8] = hi[i]; }
    return r;
}

// B-matrix 32x16 bf16 loaded from a row-major [N][K] (i.e. B^T) source:
// lane n = lane%16, khalf = lane/16; K = khalf*8 + 0..7, then 16 + khalf*8 + 0..7.
__device__ __forceinline__ bf16x16 load_b32x16(const __bf16* baseT, int ld) {
    const int l  = threadIdx.x & 31;
    const int n  = l & 15;
    const int kh = l >> 4;
    const __bf16* p = baseT + (size_t)n * ld + kh * 8;
    bf16x8 lo = *(const bf16x8*)(p);
    bf16x8 hi = *(const bf16x8*)(p + 16);
    bf16x16 r;
#pragma unroll
    for (int i = 0; i < 8; ++i) { r[i] = lo[i]; r[i + 8] = hi[i]; }
    return r;
}

// ---- TDM 2D tile load: global -> LDS, data_size = 8B units -----------------
// D# per CDNA5 ISA. Tracked with TENSORcnt.

__device__ __forceinline__ void tdm_load_2d(unsigned lds_off, const void* gaddr,
                                            unsigned tensor_d0, unsigned tensor_d1,
                                            unsigned tile_d0, unsigned tile_d1,
                                            unsigned stride0) {
    const unsigned long long ga = (unsigned long long)gaddr;
    u32x4 g0;
    g0[0] = 1u;                                   // count=1, is_restore=0
    g0[1] = lds_off;                              // lds_addr [63:32]
    g0[2] = (unsigned)ga;                         // global_addr [95:64]
    g0[3] = (unsigned)(ga >> 32) | (2u << 30);    // addr hi + type=2 ("image")
    i32x8 g1;
    g1[0] = (int)(3u << 16);                      // workgroup_mask=0, data_size=8B
    g1[1] = (int)((tensor_d0 & 0xffffu) << 16);   // tensor_dim0[15:0] @ bit48
    g1[2] = (int)((tensor_d0 >> 16) | ((tensor_d1 & 0xffffu) << 16));
    g1[3] = (int)((tensor_d1 >> 16) | (tile_d0 << 16));   // tile_dim0 @ bit112
    g1[4] = (int)(tile_d1 & 0xffffu);             // tile_dim1; tile_dim2=0
    g1[5] = (int)stride0;                         // tensor_dim0_stride[31:0]
    g1[6] = 0;                                    // stride hi, dim1_stride lo
    g1[7] = 0;
    i32x4 z4 = {0, 0, 0, 0};
#if __clang_major__ >= 23
    i32x8 z8 = {};
    __builtin_amdgcn_tensor_load_to_lds(g0, g1, z4, z4, z8, 0);
#else
    __builtin_amdgcn_tensor_load_to_lds(g0, g1, z4, z4, 0);
#endif
}

// ---- 1) LayerNorm -> bf16 --------------------------------------------------

__global__ __launch_bounds__(256) void ln_kernel(
    const float* __restrict__ x, const float* __restrict__ g,
    const float* __restrict__ b, __bf16* __restrict__ xn) {
    __shared__ float red[8];
    const int row = blockIdx.x;
    const int tid = threadIdx.x;
    const float4 v = ((const float4*)(x + (size_t)row * DIM))[tid];

    float s = v.x + v.y + v.z + v.w;
#pragma unroll
    for (int m = 1; m < 32; m <<= 1) s += __shfl_xor(s, m, 32);
    if ((tid & 31) == 0) red[tid >> 5] = s;
    __syncthreads();
    float tot = 0.f;
#pragma unroll
    for (int i = 0; i < 8; ++i) tot += red[i];
    const float mu = tot * (1.0f / DIM);

    const float dx = v.x - mu, dy = v.y - mu, dz = v.z - mu, dw = v.w - mu;
    float sq = dx * dx + dy * dy + dz * dz + dw * dw;
#pragma unroll
    for (int m = 1; m < 32; m <<= 1) sq += __shfl_xor(sq, m, 32);
    __syncthreads();
    if ((tid & 31) == 0) red[tid >> 5] = sq;
    __syncthreads();
    float vr = 0.f;
#pragma unroll
    for (int i = 0; i < 8; ++i) vr += red[i];
    const float rs = rsqrtf(vr * (1.0f / DIM) + 1e-6f);

    const int c = tid * 4;
    bf16x4 o;
    o[0] = (__bf16)(dx * rs * g[c + 0] + b[c + 0]);
    o[1] = (__bf16)(dy * rs * g[c + 1] + b[c + 1]);
    o[2] = (__bf16)(dz * rs * g[c + 2] + b[c + 2]);
    o[3] = (__bf16)(dw * rs * g[c + 3] + b[c + 3]);
    *(bf16x4*)(xn + (size_t)row * DIM + c) = o;
}

// ---- 2) fp32 -> bf16 transpose (weights -> [N][K]) -------------------------

__global__ __launch_bounds__(256) void transpose_to_bf16(
    const float* __restrict__ src, __bf16* __restrict__ dst, int K, int N) {
    const int gid = blockIdx.x * 256 + threadIdx.x;
    if (gid < K * N) {
        const int c = gid / K;
        const int k = gid - c * K;
        dst[gid] = (__bf16)src[(size_t)k * N + c];
    }
}

// ---- 3) QKV GEMM (pipelined, M=32 per wave) --------------------------------

__global__ __launch_bounds__(32) void qkv_gemm(
    const __bf16* __restrict__ xn, const __bf16* __restrict__ wT,
    __bf16* __restrict__ q, __bf16* __restrict__ k, __bf16* __restrict__ vt) {
    const int i0 = blockIdx.x * 32;
    const int c0 = blockIdx.y * 64;
    f32x8 acc0[4] = {}, acc1[4] = {};
    const __bf16* arow0 = xn + (size_t)i0 * DIM;
    const __bf16* arow1 = arow0 + (size_t)16 * DIM;

    bf16x16 a0 = load_a16x32(arow0, DIM);
    bf16x16 a1 = load_a16x32(arow1, DIM);
    bf16x16 b0 = load_b32x16(wT + (size_t)(c0 +  0) * DIM, DIM);
    bf16x16 b1 = load_b32x16(wT + (size_t)(c0 + 16) * DIM, DIM);
    bf16x16 b2 = load_b32x16(wT + (size_t)(c0 + 32) * DIM, DIM);
    bf16x16 b3 = load_b32x16(wT + (size_t)(c0 + 48) * DIM, DIM);

    for (int k0 = 0; k0 < DIM; k0 += 32) {
        const int kn = k0 + 32;
        bf16x16 a0n, a1n, b0n, b1n, b2n, b3n;
        if (kn < DIM) {     // prefetch next K-step while WMMAs run on current
            a0n = load_a16x32(arow0 + kn, DIM);
            a1n = load_a16x32(arow1 + kn, DIM);
            b0n = load_b32x16(wT + (size_t)(c0 +  0) * DIM + kn, DIM);
            b1n = load_b32x16(wT + (size_t)(c0 + 16) * DIM + kn, DIM);
            b2n = load_b32x16(wT + (size_t)(c0 + 32) * DIM + kn, DIM);
            b3n = load_b32x16(wT + (size_t)(c0 + 48) * DIM + kn, DIM);
        }
        acc0[0] = wmma_bf16(a0, b0, acc0[0]);
        acc0[1] = wmma_bf16(a0, b1, acc0[1]);
        acc0[2] = wmma_bf16(a0, b2, acc0[2]);
        acc0[3] = wmma_bf16(a0, b3, acc0[3]);
        acc1[0] = wmma_bf16(a1, b0, acc1[0]);
        acc1[1] = wmma_bf16(a1, b1, acc1[1]);
        acc1[2] = wmma_bf16(a1, b2, acc1[2]);
        acc1[3] = wmma_bf16(a1, b3, acc1[3]);
        if (kn < DIM) {
            a0 = a0n; a1 = a1n; b0 = b0n; b1 = b1n; b2 = b2n; b3 = b3n;
        }
    }

    const int l = threadIdx.x & 31, n = l & 15, kh = l >> 4;
    const int seg = c0 >> 10;          // 0=Q 1=K 2=V
    const int h   = (c0 & 1023) >> 6;  // head
#pragma unroll
    for (int ri = 0; ri < 2; ++ri) {
        const f32x8* acc = ri ? acc1 : acc0;
        const int ib = i0 + 16 * ri;
#pragma unroll
        for (int t = 0; t < 4; ++t) {
            const int d = t * 16 + n;
            if (seg == 2) {            // V^T [h][d][n]: 8 consecutive tokens
                bf16x8 pk;
#pragma unroll
                for (int r = 0; r < 8; ++r) pk[r] = (__bf16)acc[t][r];
                *(bf16x8*)(vt + (size_t)h * DHEAD * N_TOK + (size_t)d * N_TOK +
                           ib + kh * 8) = pk;
            } else {                   // Q/K row-major [h][n][d]
                __bf16* dst = (seg == 0) ? q : k;
#pragma unroll
                for (int r = 0; r < 8; ++r) {
                    const int i = ib + r + 8 * kh;
                    dst[(size_t)h * N_TOK * DHEAD + (size_t)i * DHEAD + d] =
                        (__bf16)acc[t][r];
                }
            }
        }
    }
}

// ---- 4) Flash attention: TDM double-buffered K/V tiles in LDS --------------

__global__ __launch_bounds__(128) void attn_kernel(
    const __bf16* __restrict__ q, const __bf16* __restrict__ k,
    const __bf16* __restrict__ vt, __bf16* __restrict__ obf) {
    __shared__ __align__(16) __bf16 kbuf[2][32][DHEAD];  // 32 tokens x 64 d
    __shared__ __align__(16) __bf16 vbuf[2][DHEAD][32];  // 64 d x 32 tokens
    __shared__ __align__(16) __bf16 plds[4][16][40];     // per-wave P staging

    const int w  = threadIdx.x >> 5;
    const int l  = threadIdx.x & 31;
    const int n  = l & 15;
    const int kh = l >> 4;
    const int h  = blockIdx.y;
    const int i0 = blockIdx.x * 64 + w * 16;

    const __bf16* qh  = q  + (size_t)h * N_TOK * DHEAD;
    const __bf16* khp = k  + (size_t)h * N_TOK * DHEAD;
    const __bf16* vth = vt + (size_t)h * DHEAD * N_TOK;

    const bf16x16 qa0 = load_a16x32(qh + (size_t)i0 * DHEAD,      DHEAD);
    const bf16x16 qa1 = load_a16x32(qh + (size_t)i0 * DHEAD + 32, DHEAD);

    f32x8 o0 = {}, o1 = {}, o2 = {}, o3 = {};
    float mrun[8], lrun[8];
#pragma unroll
    for (int r = 0; r < 8; ++r) { mrun[r] = -1e30f; lrun[r] = 0.f; }
    const float scale = 0.03125f;  // 1024^-0.5

    // prologue: TDM K/V tiles for j0=0 into buffer 0
    if (w == 0) {
        tdm_load_2d((unsigned)(uintptr_t)&kbuf[0][0][0], khp,
                    /*td0=*/DHEAD * 2 / 8, /*td1=*/N_TOK,
                    /*tile0=*/DHEAD * 2 / 8, /*tile1=*/32,
                    /*stride0=*/DHEAD * 2 / 8);
        tdm_load_2d((unsigned)(uintptr_t)&vbuf[0][0][0], vth,
                    /*td0=*/N_TOK * 2 / 8, /*td1=*/DHEAD,
                    /*tile0=*/32 * 2 / 8, /*tile1=*/DHEAD,
                    /*stride0=*/N_TOK * 2 / 8);
        __builtin_amdgcn_s_wait_tensorcnt(0);
    }
    __syncthreads();

    for (int j0 = 0; j0 < N_TOK; j0 += 32) {
        const int cur = (j0 >> 5) & 1;
        const int nxt = cur ^ 1;
        const bool more = (j0 + 32) < N_TOK;
        if (w == 0 && more) {  // DMA next tiles while this one is consumed
            tdm_load_2d((unsigned)(uintptr_t)&kbuf[nxt][0][0],
                        khp + (size_t)(j0 + 32) * DHEAD,
                        DHEAD * 2 / 8, N_TOK, DHEAD * 2 / 8, 32, DHEAD * 2 / 8);
            tdm_load_2d((unsigned)(uintptr_t)&vbuf[nxt][0][0],
                        vth + (j0 + 32),
                        N_TOK * 2 / 8, DHEAD, 32 * 2 / 8, DHEAD, N_TOK * 2 / 8);
        }

        // S = Q K^T from LDS K tile
        f32x8 s0 = {}, s1 = {};
        s0 = wmma_bf16(qa0, load_b32x16(&kbuf[cur][ 0][ 0], DHEAD), s0);
        s0 = wmma_bf16(qa1, load_b32x16(&kbuf[cur][ 0][32], DHEAD), s0);
        s1 = wmma_bf16(qa0, load_b32x16(&kbuf[cur][16][ 0], DHEAD), s1);
        s1 = wmma_bf16(qa1, load_b32x16(&kbuf[cur][16][32], DHEAD), s1);

        // online softmax over the 32 new columns
#pragma unroll
        for (int r = 0; r < 8; ++r) {
            const float a0 = s0[r] * scale;
            const float a1 = s1[r] * scale;
            float mx = fmaxf(a0, a1);
#pragma unroll
            for (int msk = 1; msk < 16; msk <<= 1)
                mx = fmaxf(mx, __shfl_xor(mx, msk, 32));
            const float mnew = fmaxf(mrun[r], mx);
            const float p0   = __expf(a0 - mnew);
            const float p1   = __expf(a1 - mnew);
            const float corr = __expf(mrun[r] - mnew);
            mrun[r] = mnew;
            float ps = p0 + p1;
#pragma unroll
            for (int msk = 1; msk < 16; msk <<= 1)
                ps += __shfl_xor(ps, msk, 32);
            lrun[r] = lrun[r] * corr + ps;
            o0[r] *= corr; o1[r] *= corr; o2[r] *= corr; o3[r] *= corr;
            const int m = r + 8 * kh;
            plds[w][m][n]      = (__bf16)p0;
            plds[w][m][16 + n] = (__bf16)p1;
        }
        asm volatile("s_wait_dscnt 0" ::: "memory");

        // reload P in WMMA A layout (wave-private LDS region)
        bf16x16 pa;
        {
            const __bf16* p = &plds[w][n][kh * 8];
            bf16x8 lo = *(const bf16x8*)(p);
            bf16x8 hi = *(const bf16x8*)(p + 16);
#pragma unroll
            for (int i2 = 0; i2 < 8; ++i2) { pa[i2] = lo[i2]; pa[i2 + 8] = hi[i2]; }
        }
        // O += P V from LDS V^T tile
        o0 = wmma_bf16(pa, load_b32x16(&vbuf[cur][ 0][0], 32), o0);
        o1 = wmma_bf16(pa, load_b32x16(&vbuf[cur][16][0], 32), o1);
        o2 = wmma_bf16(pa, load_b32x16(&vbuf[cur][32][0], 32), o2);
        o3 = wmma_bf16(pa, load_b32x16(&vbuf[cur][48][0], 32), o3);

        __syncthreads();               // all waves done reading buffers[cur]
        if (w == 0 && more) __builtin_amdgcn_s_wait_tensorcnt(0);
        __syncthreads();               // buffers[nxt] visible to all waves
    }

    // finalize: divide by l, store bf16 O row-major [n][h*64+d]
#pragma unroll
    for (int r = 0; r < 8; ++r) {
        const float inv = 1.0f / lrun[r];
        const int i = i0 + r + 8 * kh;
        __bf16* dst = obf + (size_t)i * DIM + h * DHEAD;
        dst[ 0 + n] = (__bf16)(o0[r] * inv);
        dst[16 + n] = (__bf16)(o1[r] * inv);
        dst[32 + n] = (__bf16)(o2[r] * inv);
        dst[48 + n] = (__bf16)(o3[r] * inv);
    }
}

// ---- 5) Output projection (pipelined, M=32 per wave) -----------------------

__global__ __launch_bounds__(32) void out_gemm(
    const __bf16* __restrict__ obf, const __bf16* __restrict__ woT,
    const float* __restrict__ bout, float* __restrict__ out) {
    const int i0 = blockIdx.x * 32;
    const int c0 = blockIdx.y * 64;
    f32x8 acc0[4] = {}, acc1[4] = {};
    const __bf16* arow0 = obf + (size_t)i0 * DIM;
    const __bf16* arow1 = arow0 + (size_t)16 * DIM;

    bf16x16 a0 = load_a16x32(arow0, DIM);
    bf16x16 a1 = load_a16x32(arow1, DIM);
    bf16x16 b0 = load_b32x16(woT + (size_t)(c0 +  0) * DIM, DIM);
    bf16x16 b1 = load_b32x16(woT + (size_t)(c0 + 16) * DIM, DIM);
    bf16x16 b2 = load_b32x16(woT + (size_t)(c0 + 32) * DIM, DIM);
    bf16x16 b3 = load_b32x16(woT + (size_t)(c0 + 48) * DIM, DIM);

    for (int k0 = 0; k0 < DIM; k0 += 32) {
        const int kn = k0 + 32;
        bf16x16 a0n, a1n, b0n, b1n, b2n, b3n;
        if (kn < DIM) {
            a0n = load_a16x32(arow0 + kn, DIM);
            a1n = load_a16x32(arow1 + kn, DIM);
            b0n = load_b32x16(woT + (size_t)(c0 +  0) * DIM + kn, DIM);
            b1n = load_b32x16(woT + (size_t)(c0 + 16) * DIM + kn, DIM);
            b2n = load_b32x16(woT + (size_t)(c0 + 32) * DIM + kn, DIM);
            b3n = load_b32x16(woT + (size_t)(c0 + 48) * DIM + kn, DIM);
        }
        acc0[0] = wmma_bf16(a0, b0, acc0[0]);
        acc0[1] = wmma_bf16(a0, b1, acc0[1]);
        acc0[2] = wmma_bf16(a0, b2, acc0[2]);
        acc0[3] = wmma_bf16(a0, b3, acc0[3]);
        acc1[0] = wmma_bf16(a1, b0, acc1[0]);
        acc1[1] = wmma_bf16(a1, b1, acc1[1]);
        acc1[2] = wmma_bf16(a1, b2, acc1[2]);
        acc1[3] = wmma_bf16(a1, b3, acc1[3]);
        if (kn < DIM) {
            a0 = a0n; a1 = a1n; b0 = b0n; b1 = b1n; b2 = b2n; b3 = b3n;
        }
    }

    const int l = threadIdx.x & 31, n = l & 15, kh = l >> 4;
#pragma unroll
    for (int ri = 0; ri < 2; ++ri) {
        const f32x8* acc = ri ? acc1 : acc0;
        const int ib = i0 + 16 * ri;
#pragma unroll
        for (int t = 0; t < 4; ++t) {
            const int c = c0 + t * 16 + n;
            const float bb = bout[c];
#pragma unroll
            for (int r = 0; r < 8; ++r) {
                const int i = ib + r + 8 * kh;
                out[(size_t)i * DIM + c] = acc[t][r] + bb;
            }
        }
    }
}

// ---- launch ----------------------------------------------------------------

extern "C" void kernel_launch(void* const* d_in, const int* in_sizes, int n_in,
                              void* d_out, int out_size, void* d_ws, size_t ws_size,
                              hipStream_t stream) {
    const float* x    = (const float*)d_in[0];
    const float* lns  = (const float*)d_in[1];
    const float* lnb  = (const float*)d_in[2];
    const float* wqkv = (const float*)d_in[3];
    const float* wout = (const float*)d_in[4];
    const float* bout = (const float*)d_in[5];
    float* out = (float*)d_out;

    __bf16* xn    = (__bf16*)d_ws;                     // 2048*1024
    __bf16* wqkvT = xn    + (size_t)N_TOK * DIM;       // 3072*1024  ([N][K])
    __bf16* woutT = wqkvT + (size_t)3 * DIM * DIM;     // 1024*1024  ([N][K])
    __bf16* qb    = woutT + (size_t)DIM * DIM;         // [h][n][d]
    __bf16* kb    = qb    + (size_t)N_TOK * DIM;       // [h][n][d]
    __bf16* vtb   = kb    + (size_t)N_TOK * DIM;       // [h][d][n]
    __bf16* ob    = vtb   + (size_t)N_TOK * DIM;       // [n][h*d]

    ln_kernel<<<N_TOK, 256, 0, stream>>>(x, lns, lnb, xn);
    transpose_to_bf16<<<(3 * DIM * DIM) / 256, 256, 0, stream>>>(wqkv, wqkvT, DIM, 3 * DIM);
    transpose_to_bf16<<<(DIM * DIM) / 256, 256, 0, stream>>>(wout, woutT, DIM, DIM);
    qkv_gemm<<<dim3(N_TOK / 32, (3 * DIM) / 64), 32, 0, stream>>>(xn, wqkvT, qb, kb, vtb);
    attn_kernel<<<dim3(N_TOK / 64, HEADS), 128, 0, stream>>>(qb, kb, vtb, ob);
    out_gemm<<<dim3(N_TOK / 32, DIM / 64), 32, 0, stream>>>(ob, woutT, bout, out);
}
